// KnittingPatternMemory_35682588295781
// MI455X (gfx1250) — matrix operand order repaired and checked by reading, not compiled, not run on hardware
//
#include <hip/hip_runtime.h>
#include <hip/hip_bf16.h>
#include <math.h>

// ---------------------------------------------------------------------------
// KnittingPatternMemory on MI455X (gfx1250): bf16 WMMA mixed-precision pipeline
// Block tile 128x128 (256 thr / 8 waves), wave tile 32x64, BK=32, double-
// buffered LDS, vectorized fp32->bf16 staging, branchless edge guards.
// ---------------------------------------------------------------------------

typedef __attribute__((ext_vector_type(16))) __bf16 v16bf;
typedef __attribute__((ext_vector_type(8)))  __bf16 v8bf;
typedef __attribute__((ext_vector_type(4)))  __bf16 v4bf;
typedef __attribute__((ext_vector_type(8)))  float  v8f;
typedef __attribute__((ext_vector_type(4)))  float  v4f;

#define NSEG   10
#define LSEQ   680            // sum of p^2 over PATCH_NUMS
#define CDIM   1024
#define BATCH  64
#define NKEY_TOT 1280         // NS * S
#define MQ_TOT 43520          // BATCH * LSEQ
#define PROBS_PER_B 738560ll  // sum l_i * N_i

__device__ const int d_seg_start[NSEG] = {0,1,5,14,30,55,91,155,255,424};
__device__ const int d_seg_len[NSEG]   = {1,4,9,16,25,36,64,100,169,256};
__device__ const long long d_cum_sc[NSEG] =
    {0,128,1152,4608,12800,28800,56448,113792,216192,410880};

// Row stride 40 bf16 = 80B: 16B-aligned fragment chunks, odd bank stride.
#define TSTRIDE 40

// ---------------------------------------------------------------------------
// Core 128x128 tile GEMM, BK=32, 8 waves, bf16 WMMA, fp32 accumulate.
//   A: MxK row-major fp32 (rows guarded vs M)
//   B: TRANSB==0 -> KxN row-major; TRANSB==1 -> NxK row-major (uses B^T)
//   C[row,col] = alpha * (acc + bias[col])
// Fragment layouts per CDNA5 ISA 7.12.2 (wave32):
//   A frag 16x32: M=lane&15, kh=lane>>4, K = 8*kh + (e<8?e:e+8)
//   B frag 32x16: N=lane&15, K = 8*kh + (e<8?e:e+8)  (LDS stored as [n][k])
//   D 16x16 f32:  N=lane&15, M = r + 8*kh
// ---------------------------------------------------------------------------
template <int TRANSB>
__device__ __forceinline__ void gemm_core(
    const float* __restrict__ A, int lda, int M, int row0,
    const float* __restrict__ Bm, int ldb, int col0, int N,
    int K,
    float* __restrict__ C, int ldc,
    const float* __restrict__ bias, float alpha)
{
  __shared__ __align__(16) __bf16 As[2][128 * TSTRIDE];
  __shared__ __align__(16) __bf16 Bs[2][128 * TSTRIDE];

  const int tid  = threadIdx.x;       // 0..255
  const int w    = tid >> 5;          // wave 0..7
  const int lane = tid & 31;
  const int nrow = lane & 15;
  const int kh   = lane >> 4;
  const int wr   = w & 3;             // row group: rows wr*32 .. +32
  const int wc   = w >> 2;            // col group: cols wc*64 .. +64

  // staging maps
  const int ac = tid & 7;             // k-chunk (A tile / B^T tile)
  const int am = tid >> 3;            // base row 0..31 (rows am + 32*it)
  const int bn = tid & 31;            // n-block (B no-trans)
  const int bk = tid >> 5;            // k-block 0..7 (B no-trans)

  v4f areg[4], breg[4];
  v8f acc[2][4];
  #pragma unroll
  for (int u = 0; u < 2; ++u)
    #pragma unroll
    for (int t = 0; t < 4; ++t) { v8f z = {}; acc[u][t] = z; }

  // ---- fetch tile (kk) from global into registers, branchless guards ----
  auto fetch = [&](int kk) {
    #pragma unroll
    for (int it = 0; it < 4; ++it) {
      int gr = row0 + am + 32 * it;
      int cr = (gr < M) ? gr : (M - 1);
      float s = (gr < M) ? 1.0f : 0.0f;
      v4f v = *(const v4f*)&A[(size_t)cr * lda + kk + 4 * ac];
      areg[it] = v * s;
    }
    if (TRANSB) {
      #pragma unroll
      for (int it = 0; it < 4; ++it) {
        int gn = col0 + am + 32 * it;
        int cn = (gn < N) ? gn : (N - 1);
        float s = (gn < N) ? 1.0f : 0.0f;
        v4f v = *(const v4f*)&Bm[(size_t)cn * ldb + kk + 4 * ac];
        breg[it] = v * s;
      }
    } else {
      int gc = col0 + 4 * bn;                 // N is a multiple of 4
      int cc = (gc < N) ? gc : (N - 4);
      float s = (gc < N) ? 1.0f : 0.0f;
      #pragma unroll
      for (int j = 0; j < 4; ++j) {
        v4f v = *(const v4f*)&Bm[(size_t)(kk + 4 * bk + j) * ldb + cc];
        breg[j] = v * s;
      }
    }
  };

  // ---- convert + store registers into LDS buffer ----
  auto stage = [&](int buf) {
    #pragma unroll
    for (int it = 0; it < 4; ++it) {
      v4bf t = {(__bf16)areg[it][0], (__bf16)areg[it][1],
                (__bf16)areg[it][2], (__bf16)areg[it][3]};
      *(v4bf*)&As[buf][(am + 32 * it) * TSTRIDE + 4 * ac] = t;
    }
    if (TRANSB) {
      #pragma unroll
      for (int it = 0; it < 4; ++it) {
        v4bf t = {(__bf16)breg[it][0], (__bf16)breg[it][1],
                  (__bf16)breg[it][2], (__bf16)breg[it][3]};
        *(v4bf*)&Bs[buf][(am + 32 * it) * TSTRIDE + 4 * ac] = t;
      }
    } else {
      #pragma unroll
      for (int j2 = 0; j2 < 4; ++j2) {        // 4x4 micro-transpose -> [n][k]
        v4bf t = {(__bf16)breg[0][j2], (__bf16)breg[1][j2],
                  (__bf16)breg[2][j2], (__bf16)breg[3][j2]};
        *(v4bf*)&Bs[buf][(4 * bn + j2) * TSTRIDE + 4 * bk] = t;
      }
    }
  };

  auto frag_a = [&](int buf, int u) -> v16bf {
    const __bf16* p = &As[buf][(wr * 32 + u * 16 + nrow) * TSTRIDE];
    v8bf lo = *(const v8bf*)(p + 8 * kh);
    v8bf hi = *(const v8bf*)(p + 16 + 8 * kh);
    return __builtin_shufflevector(lo, hi,
        0,1,2,3,4,5,6,7,8,9,10,11,12,13,14,15);
  };
  auto frag_b = [&](int buf, int t) -> v16bf {
    const __bf16* p = &Bs[buf][(wc * 64 + t * 16 + nrow) * TSTRIDE];
    v8bf lo = *(const v8bf*)(p + 8 * kh);
    v8bf hi = *(const v8bf*)(p + 16 + 8 * kh);
    return __builtin_shufflevector(lo, hi,
        0,1,2,3,4,5,6,7,8,9,10,11,12,13,14,15);
  };

  // ---- software pipeline: fetch(k+1) overlaps WMMA(k) ----
  fetch(0);
  stage(0);
  __syncthreads();
  int buf = 0;
  for (int kk = 0; kk < K; kk += 32) {
    const bool more = (kk + 32) < K;
    if (more) fetch(kk + 32);               // global loads in flight
    v16bf a0 = frag_a(buf, 0);
    v16bf a1 = frag_a(buf, 1);
    #pragma unroll
    for (int t = 0; t < 4; ++t) {
      v16bf bt = frag_b(buf, t);
      acc[0][t] = __builtin_amdgcn_wmma_f32_16x16x32_bf16(
          false, a0, false, bt, (short)0, acc[0][t], false, false);
      acc[1][t] = __builtin_amdgcn_wmma_f32_16x16x32_bf16(
          false, a1, false, bt, (short)0, acc[1][t], false, false);
    }
    if (more) stage(buf ^ 1);               // writes other buffer: no race
    __syncthreads();
    buf ^= 1;
  }

  // ---- epilogue: C = alpha * (acc + bias[col]) ----
  #pragma unroll
  for (int t = 0; t < 4; ++t) {
    int col = col0 + wc * 64 + t * 16 + nrow;
    float bv = (bias != nullptr && col < N) ? bias[col] : 0.0f;
    #pragma unroll
    for (int u = 0; u < 2; ++u) {
      #pragma unroll
      for (int r = 0; r < 8; ++r) {
        int row = row0 + wr * 32 + u * 16 + r + 8 * kh;
        if (row < M && col < N)
          C[(size_t)row * ldc + col] = alpha * (acc[u][t][r] + bv);
      }
    }
  }
}

// ---------------------------------------------------------------------------
// Plain GEMM: alpha_mode 0 -> 1.0 ; 2 -> sigmoid(*gate_ptr)
// ---------------------------------------------------------------------------
__global__ void __launch_bounds__(256)
k_gemm_plain(const float* __restrict__ A, const float* __restrict__ Bm,
             float* __restrict__ C, const float* __restrict__ bias,
             const float* __restrict__ gate_ptr,
             int M, int N, int K, int lda, int ldb, int ldc, int alpha_mode)
{
  int row0 = blockIdx.y * 128;
  int col0 = blockIdx.x * 128;
  if (row0 >= M || col0 >= N) return;
  float alpha = 1.0f;
  if (alpha_mode == 2) alpha = 1.0f / (1.0f + expf(-gate_ptr[0]));
  gemm_core<0>(A, lda, M, row0, Bm, ldb, col0, N, K, C, ldc, bias, alpha);
}

// ---------------------------------------------------------------------------
// Segmented scores: probs(b,i) = (Q_seg @ K_vis^T) * inv_scale / temp
// ---------------------------------------------------------------------------
__global__ void __launch_bounds__(256)
k_scores(const float* __restrict__ query, const float* __restrict__ keys,
         float* __restrict__ probs, const float* __restrict__ log_temp)
{
  int z = blockIdx.z;
  int b = z / NSEG, i = z - b * NSEG;
  int M = d_seg_len[i];
  int N = 128 * (i + 1);
  int row0 = blockIdx.y * 128;
  int col0 = blockIdx.x * 128;
  if (row0 >= M || col0 >= N) return;
  float temp = expf(log_temp[0]);
  temp = fminf(fmaxf(temp, 0.05f), 1.0f);
  float alpha = 0.03125f / temp;            // 1/sqrt(1024) / temp
  const float* A = query + ((size_t)b * LSEQ + d_seg_start[i]) * CDIM;
  float* C = probs + (size_t)b * PROBS_PER_B + d_cum_sc[i];
  gemm_core<1>(A, CDIM, M, row0, keys, CDIM, col0, N, CDIM, C, N,
               nullptr, alpha);
}

// ---------------------------------------------------------------------------
// Row softmax over scores, fp32, one 256-thread block per (b, row)
// ---------------------------------------------------------------------------
__global__ void __launch_bounds__(256)
k_softmax(float* __restrict__ probs)
{
  __shared__ float red[256];
  int r  = blockIdx.x;
  int b  = r / LSEQ;
  int lr = r - b * LSEQ;
  int i = 0;
  #pragma unroll
  for (int s = 0; s < NSEG; ++s) if (lr >= d_seg_start[s]) i = s;
  int Nvis = 128 * (i + 1);
  float* row = probs + (size_t)b * PROBS_PER_B + d_cum_sc[i] +
               (size_t)(lr - d_seg_start[i]) * Nvis;
  int tid = threadIdx.x;

  float m = -3.4e38f;
  for (int j = tid; j < Nvis; j += 256) m = fmaxf(m, row[j]);
  red[tid] = m; __syncthreads();
  for (int s = 128; s > 0; s >>= 1) {
    if (tid < s) red[tid] = fmaxf(red[tid], red[tid + s]);
    __syncthreads();
  }
  m = red[0]; __syncthreads();

  float acc = 0.0f;
  for (int j = tid; j < Nvis; j += 256) {
    float e = expf(row[j] - m);
    row[j] = e;
    acc += e;
  }
  red[tid] = acc; __syncthreads();
  for (int s = 128; s > 0; s >>= 1) {
    if (tid < s) red[tid] += red[tid + s];
    __syncthreads();
  }
  float inv = 1.0f / red[0];
  for (int j = tid; j < Nvis; j += 256) row[j] *= inv;
}

// ---------------------------------------------------------------------------
// Segmented PV: mem_combined_seg = attn @ V_vis
// ---------------------------------------------------------------------------
__global__ void __launch_bounds__(256)
k_pv(const float* __restrict__ probs, const float* __restrict__ values,
     float* __restrict__ memc)
{
  int z = blockIdx.z;
  int b = z / NSEG, i = z - b * NSEG;
  int M = d_seg_len[i];
  int Nvis = 128 * (i + 1);
  int row0 = blockIdx.y * 128;
  int col0 = blockIdx.x * 128;              // N = 1024 exact
  if (row0 >= M) return;
  const float* A = probs + (size_t)b * PROBS_PER_B + d_cum_sc[i];
  float* C = memc + ((size_t)b * LSEQ + d_seg_start[i]) * CDIM;
  gemm_core<0>(A, Nvis, M, row0, values, CDIM, col0, CDIM, Nvis,
               C, CDIM, nullptr, 1.0f);
}

__global__ void k_tail_zero(float* __restrict__ out, unsigned long long off)
{
  if (threadIdx.x < 2) out[off + threadIdx.x] = 0.0f;
}

// ---------------------------------------------------------------------------
extern "C" void kernel_launch(void* const* d_in, const int* in_sizes, int n_in,
                              void* d_out, int out_size, void* d_ws, size_t ws_size,
                              hipStream_t stream) {
  (void)in_sizes; (void)n_in; (void)out_size; (void)ws_size;
  const float* x    = (const float*)d_in[0];
  const float* mem  = (const float*)d_in[1];
  const float* Wq   = (const float*)d_in[2];
  const float* Wk   = (const float*)d_in[3];
  const float* Wv   = (const float*)d_in[4];
  const float* wk1  = (const float*)d_in[5];
  const float* bk1  = (const float*)d_in[6];
  const float* wk2  = (const float*)d_in[7];
  const float* bk2  = (const float*)d_in[8];
  const float* wv1  = (const float*)d_in[9];
  const float* bv1  = (const float*)d_in[10];
  const float* wv2  = (const float*)d_in[11];
  const float* bv2  = (const float*)d_in[12];
  const float* gk   = (const float*)d_in[13];
  const float* gv   = (const float*)d_in[14];
  const float* ltmp = (const float*)d_in[15];
  // d_in[16] = begin_ends: compile-time constants (PATCH_NUMS)

  float* ws = (float*)d_ws;
  size_t o = 0;
  float* query  = ws + o; o += (size_t)MQ_TOT * CDIM;
  float* keys   = ws + o; o += (size_t)NKEY_TOT * CDIM;
  float* values = ws + o; o += (size_t)NKEY_TOT * CDIM;
  float* probs  = ws + o; o += (size_t)BATCH * PROBS_PER_B;
  float* memc   = ws + o; o += (size_t)MQ_TOT * CDIM;
  float* tlow   = ws + o;                       // 43520x64, reused k then v

  float* outk = (float*)d_out;
  float* outv = outk + (size_t)MQ_TOT * CDIM;

  dim3 blk(256);

  // 1) query = x @ Wq
  k_gemm_plain<<<dim3(8, 340, 1), blk, 0, stream>>>(
      x, Wq, query, nullptr, nullptr, MQ_TOT, 1024, 1024, 1024, 1024, 1024, 0);
  // 2) keys / values = mem @ Wk / Wv
  k_gemm_plain<<<dim3(8, 10, 1), blk, 0, stream>>>(
      mem, Wk, keys, nullptr, nullptr, NKEY_TOT, 1024, 1024, 1024, 1024, 1024, 0);
  k_gemm_plain<<<dim3(8, 10, 1), blk, 0, stream>>>(
      mem, Wv, values, nullptr, nullptr, NKEY_TOT, 1024, 1024, 1024, 1024, 1024, 0);
  // 3) scores, 4) softmax, 5) attn @ V
  k_scores<<<dim3(10, 2, BATCH * NSEG), blk, 0, stream>>>(query, keys, probs, ltmp);
  k_softmax<<<dim3(MQ_TOT), dim3(256), 0, stream>>>(probs);
  k_pv<<<dim3(8, 2, BATCH * NSEG), blk, 0, stream>>>(probs, values, memc);
  // 6) mem_k = sigmoid(gk) * ((memc @ wk1 + bk1) @ wk2 + bk2)
  k_gemm_plain<<<dim3(1, 340, 1), blk, 0, stream>>>(
      memc, wk1, tlow, bk1, nullptr, MQ_TOT, 64, 1024, 1024, 64, 64, 0);
  k_gemm_plain<<<dim3(8, 340, 1), blk, 0, stream>>>(
      tlow, wk2, outk, bk2, gk, MQ_TOT, 1024, 64, 64, 1024, 1024, 2);
  // 7) mem_v branch
  k_gemm_plain<<<dim3(1, 340, 1), blk, 0, stream>>>(
      memc, wv1, tlow, bv1, nullptr, MQ_TOT, 64, 1024, 1024, 64, 64, 0);
  k_gemm_plain<<<dim3(8, 340, 1), blk, 0, stream>>>(
      tlow, wv2, outv, bv2, gv, MQ_TOT, 1024, 64, 64, 1024, 1024, 2);
  // 8) trailing scalar zeros
  k_tail_zero<<<1, 32, 0, stream>>>((float*)d_out,
                                    (unsigned long long)2 * MQ_TOT * CDIM);
}
